// HierarchicalRouter_43688407335204
// MI455X (gfx1250) — matrix operand order repaired and checked by reading, not compiled, not run on hardware
//
#include <hip/hip_runtime.h>
#include <hip/hip_bf16.h>

#define N_TOK 32768
#define D_DIM 2048
#define G_GRP 8
#define EG    8
#define NE    64   // G*E_G
#define NC    72   // 8 gate cols + 64 expert cols
#define NCP   80   // padded to 5 x 16-col WMMA tiles
#define ROWS_PER_BLOCK 128
#define THREADS 128          // 4 waves; each wave does two 16-row M-tiles

typedef __attribute__((ext_vector_type(16))) __bf16 v16bf;
typedef __attribute__((ext_vector_type(8)))  __bf16 v8bf;
typedef __attribute__((ext_vector_type(8)))  float  v8f;
typedef __attribute__((ext_vector_type(4)))  float  v4f;

// Pack [Wg | We | zero-pad] (row-major [D, cols]) into column-major bf16
// Wpack[col][k], col = 0..79.  320 KB, L2-resident for the whole GEMM.
__global__ void hr_pack_weights(const float* __restrict__ Wg,
                                const float* __restrict__ We,
                                __bf16* __restrict__ wp) {
    int idx = blockIdx.x * blockDim.x + threadIdx.x;
    if (idx >= NCP * D_DIM) return;
    int c = idx / D_DIM;
    int k = idx % D_DIM;
    float v = 0.0f;
    if (c < G_GRP)      v = Wg[(size_t)k * G_GRP + c];
    else if (c < NC)    v = We[(size_t)k * NE + (c - G_GRP)];
    wp[(size_t)c * D_DIM + k] = (__bf16)v;
}

// Load 32 consecutive K-values of x (f32) for one 16x32 bf16 A tile,
// following the CDNA5 16-bit A layout: lanes 0-15 hold K 0..7 & 16..23,
// lanes 16-31 hold K 8..15 & 24..31 (khiA handles the split).
__device__ __forceinline__ v16bf load_a_tile(const float* __restrict__ xrow,
                                             int k0, int khiA) {
    v4f f0a = *(const v4f*)(xrow + k0 + khiA);
    v4f f0b = *(const v4f*)(xrow + k0 + khiA + 4);
    v4f f1a = *(const v4f*)(xrow + k0 + khiA + 16);
    v4f f1b = *(const v4f*)(xrow + k0 + khiA + 20);
    v16bf a;
#pragma unroll
    for (int j = 0; j < 4; ++j) {
        a[j]      = (__bf16)f0a[j];
        a[4 + j]  = (__bf16)f0b[j];
        a[8 + j]  = (__bf16)f1a[j];
        a[12 + j] = (__bf16)f1b[j];
    }
    return a;
}

__global__ __launch_bounds__(THREADS)
void hr_router_kernel(const float* __restrict__ x,
                      const __bf16* __restrict__ wp,
                      float* __restrict__ out_mask,
                      float* __restrict__ out_w) {
    __shared__ float slog[ROWS_PER_BLOCK * NCP];   // 40 KB logits staging

    const int tid  = threadIdx.x;
    const int lane = tid & 31;
    const int wave = tid >> 5;                      // 4 waves, 32 rows each
    const int rowbase = blockIdx.x * ROWS_PER_BLOCK + wave * 32;

    const int arow = lane & 15;
    const int khiA = (lane >> 4) << 3;              // A: 0 or 8
    const int kbB  = (lane >> 4) << 3;              // B: 0 or 8
    const int ncol = lane & 15;

    // acc[m][t]: m = M-tile (rows 0-15 / 16-31), t = N-tile (cols 16t..16t+15)
    v8f acc[2][5];
#pragma unroll
    for (int m = 0; m < 2; ++m)
#pragma unroll
        for (int t = 0; t < 5; ++t) acc[m][t] = (v8f){};

    const float* xrow0 = x + (size_t)(rowbase + arow) * D_DIM;
    const float* xrow1 = x + (size_t)(rowbase + 16 + arow) * D_DIM;

    for (int k0 = 0; k0 < D_DIM; k0 += 32) {
        v16bf a0 = load_a_tile(xrow0, k0, khiA);
        v16bf a1 = load_a_tile(xrow1, k0, khiA);
        // Each B tile (32x16 bf16, K 8h..8h+7 & 16+8h..23+8h per lane from the pack)
        // is amortized over both M-tiles -> halves B-side L2 traffic.
#pragma unroll
        for (int t = 0; t < 5; ++t) {
            const __bf16* wcol = wp + (size_t)(t * 16 + ncol) * D_DIM + k0 + kbB;
            v8bf b0 = *(const v8bf*)(wcol);
            v8bf b1 = *(const v8bf*)(wcol + 16);
            v16bf b;
#pragma unroll
            for (int j = 0; j < 8; ++j) { b[j] = b0[j]; b[8 + j] = b1[j]; }
            acc[0][t] = __builtin_amdgcn_wmma_f32_16x16x32_bf16(
                            false, a0, false, b, (short)0, acc[0][t], false, false);
            acc[1][t] = __builtin_amdgcn_wmma_f32_16x16x32_bf16(
                            false, a1, false, b, (short)0, acc[1][t], false, false);
        }
    }

    // C layout: VGPR r -> M = r + 8*(lane>=16), N = lane&15
    const int moff = (lane >> 4) << 3;
#pragma unroll
    for (int m = 0; m < 2; ++m) {
#pragma unroll
        for (int r = 0; r < 8; ++r) {
            float* dst = &slog[(wave * 32 + m * 16 + r + moff) * NCP + ncol];
#pragma unroll
            for (int t = 0; t < 5; ++t) dst[t * 16] = acc[m][t][r];
        }
    }
    __syncthreads();

    // ---- Routing: one thread per token (all 128 threads active) ----
    {
        float* lg = &slog[tid * NCP];

        // group softmax over 8
        float gp[G_GRP];
        float gm = -1e30f;
#pragma unroll
        for (int g = 0; g < G_GRP; ++g) gm = fmaxf(gm, lg[g]);
        float gs = 0.0f;
#pragma unroll
        for (int g = 0; g < G_GRP; ++g) { gp[g] = __expf(lg[g] - gm); gs += gp[g]; }
        float gi = 1.0f / gs;
        unsigned vlo = 0, vhi = 0;
        int cnt = 0;
#pragma unroll
        for (int g = 0; g < G_GRP; ++g) {
            gp[g] *= gi;
            bool gmask = gp[g] >= (1.0f / G_GRP);
            // expert softmax within group g; overwrite logit slots with fp
            float el[EG];
            float em = -1e30f;
#pragma unroll
            for (int e = 0; e < EG; ++e) { el[e] = lg[G_GRP + g * EG + e]; em = fmaxf(em, el[e]); }
            float es = 0.0f;
#pragma unroll
            for (int e = 0; e < EG; ++e) { el[e] = __expf(el[e] - em); es += el[e]; }
            float ei = 1.0f / es;
#pragma unroll
            for (int e = 0; e < EG; ++e) {
                float p = el[e] * ei;
                int i = g * EG + e;
                lg[G_GRP + i] = gp[g] * p;                 // fp
                bool v = gmask && (p >= (1.0f / EG));
                if (v) {
                    if (i < 32) vlo |= 1u << i; else vhi |= 1u << (i - 32);
                    ++cnt;
                }
            }
        }

        // top-2 over fp (ascending scan keeps lowest index on ties, like lax.top_k)
        float b1 = -1e30f, b2 = -1e30f;
        int i1 = -1, i2 = -1;
        for (int i = 0; i < NE; ++i) {
            float f = lg[G_GRP + i];
            if (f > b1)      { b2 = b1; i2 = i1; b1 = f; i1 = i; }
            else if (f > b2) { b2 = f; i2 = i; }
        }
        const bool useTopk = (cnt < 2);

        float sum = 0.0f;
        for (int i = 0; i < NE; ++i) {
            bool vth = (i < 32) ? ((vlo >> i) & 1u) : ((vhi >> (i - 32)) & 1u);
            bool m = useTopk ? (i == i1 || i == i2) : vth;
            if (m) sum += lg[G_GRP + i];
        }
        float inw = 1.0f / fmaxf(sum, 1e-9f);

        const size_t ob = (size_t)(blockIdx.x * ROWS_PER_BLOCK + tid) * NE;
        for (int i = 0; i < NE; ++i) {
            bool vth = (i < 32) ? ((vlo >> i) & 1u) : ((vhi >> (i - 32)) & 1u);
            bool m = useTopk ? (i == i1 || i == i2) : vth;
            out_mask[ob + i] = m ? 1.0f : 0.0f;
            out_w[ob + i]    = m ? lg[G_GRP + i] * inw : 0.0f;
        }
    }
}

extern "C" void kernel_launch(void* const* d_in, const int* in_sizes, int n_in,
                              void* d_out, int out_size, void* d_ws, size_t ws_size,
                              hipStream_t stream) {
    (void)in_sizes; (void)n_in; (void)out_size; (void)ws_size;
    const float* x  = (const float*)d_in[0];
    const float* Wg = (const float*)d_in[1];
    const float* We = (const float*)d_in[2];

    __bf16* wp = (__bf16*)d_ws;                        // 80*2048*2 B = 320 KB
    float* out_mask = (float*)d_out;                   // [N, 64] mask as 0/1
    float* out_w    = out_mask + (size_t)N_TOK * NE;   // [N, 64] weights

    hr_pack_weights<<<(NCP * D_DIM + 255) / 256, 256, 0, stream>>>(Wg, We, wp);
    hr_router_kernel<<<N_TOK / ROWS_PER_BLOCK, THREADS, 0, stream>>>(x, wp, out_mask, out_w);
}
